// GroupedQueryAttentionRoPE_90134183674453
// MI455X (gfx1250) — matrix operand order repaired and checked by reading, not compiled, not run on hardware
//
#include <hip/hip_runtime.h>
#include <hip/hip_bf16.h>
#include <math.h>
#include <utility>

// Problem constants (from reference)
#define DMODEL 1024
#define NHEADS 16
#define NKV    4
#define GRPS   4
#define DK     64
#define BB     2
#define TT     2048
#define BTOK   (BB * TT)   // 4096 token rows

// ---- WMMA fragment types (gfx1250, wave32) ----
typedef __bf16 bf16x16 __attribute__((ext_vector_type(16)));
typedef float  f32x8  __attribute__((ext_vector_type(8)));

union FragB { bf16x16 v; unsigned int u[8]; unsigned short s[16]; };
union FragC { f32x8  v; float f[8]; };

// Compile-time unrolled loop so WMMA modifier args are ICEs.
template <int... Is, typename F>
__device__ __forceinline__ void static_for_impl(F&& f, std::integer_sequence<int, Is...>) {
  (f(std::integral_constant<int, Is>{}), ...);
}
template <int N, typename F>
__device__ __forceinline__ void static_for(F&& f) {
  static_for_impl(std::forward<F>(f), std::make_integer_sequence<int, N>{});
}

__device__ __forceinline__ unsigned short f32_to_bf16_rn(float f) {
  unsigned int u = __builtin_bit_cast(unsigned int, f);
  u += 0x7FFFu + ((u >> 16) & 1u);            // round-to-nearest-even
  return (unsigned short)(u >> 16);
}
__device__ __forceinline__ float bf16_to_f32(unsigned short s) {
  unsigned int u = ((unsigned int)s) << 16;
  return __builtin_bit_cast(float, u);
}

// Async 16B global->LDS copy (per-lane), tracked by ASYNCcnt.
__device__ __forceinline__ void async_b128(unsigned lds_off, const void* gptr) {
  asm volatile("global_load_async_to_lds_b128 %0, %1, off"
               :: "v"(lds_off), "v"((unsigned long long)(uintptr_t)gptr)
               : "memory");
}
__device__ __forceinline__ void wait_async0() {
  asm volatile("s_wait_asynccnt 0x0" ::: "memory");
}

// ---------------------------------------------------------------------------
// Elementwise fp32 -> bf16 conversion
// ---------------------------------------------------------------------------
__global__ void cvt_f32_to_bf16(const float* __restrict__ in,
                                unsigned short* __restrict__ out, int n) {
  int i = blockIdx.x * blockDim.x + threadIdx.x;
  if (i < n) out[i] = f32_to_bf16_rn(in[i]);
}

// ---------------------------------------------------------------------------
// RoPE on a [B, T, nh, 64] bf16 tensor (in place). One thread per (b,t,h,i<32).
// ---------------------------------------------------------------------------
__global__ void rope_bf16(unsigned short* __restrict__ X, int T, int nh, int total) {
  int i = blockIdx.x * blockDim.x + threadIdx.x;
  if (i >= total) return;
  const int d = i & 31;
  int r = i >> 5;               // (b*T + t)*nh + h
  const size_t base = (size_t)r * 64;
  r /= nh;
  const int t = r % T;
  const float inv = __powf(10000.0f, -(float)d / 32.0f);
  float sn, cs;
  __sincosf((float)t * inv, &sn, &cs);
  const float x1 = bf16_to_f32(X[base + d]);
  const float x2 = bf16_to_f32(X[base + 32 + d]);
  X[base + d]      = f32_to_bf16_rn(x1 * cs - x2 * sn);
  X[base + 32 + d] = f32_to_bf16_rn(x2 * cs + x1 * sn);
}

// ---------------------------------------------------------------------------
// Tiled bf16 GEMM, f32 accumulate:  C[M,N] = A[M,K] * B[K,N]
// Block: 256 threads = 8 waves; WG tile 128(M) x 64(N); K staged 32 deep.
// Each wave owns a 16x64 strip: 1 A-fragment load feeds 4 WMMAs (reuse-A hint).
// A tile staged with async global->LDS b128; B staged synchronously (needs
// transpose into sBt[n][k] so B-fragment gathers are contiguous-k).
// ---------------------------------------------------------------------------
__launch_bounds__(256)
__global__ void gemm_bf16_wmma(const unsigned short* __restrict__ A,
                               const unsigned short* __restrict__ B,
                               unsigned short* __restrict__ Cb,
                               float* __restrict__ Cf,
                               int M, int N, int K) {
  __shared__ unsigned short sA[128][40];   // [m][k], 80B rows (16B aligned)
  __shared__ unsigned short sBt[64][40];   // [n][k]

  const int tid = threadIdx.x;
  const int w  = tid >> 5;
  const int l  = tid & 31;
  const int lo = l & 15, hi = l >> 4;
  const int m0 = blockIdx.y * 128, n0 = blockIdx.x * 64;

  FragC acc[4];
  #pragma unroll
  for (int nt = 0; nt < 4; ++nt)
    #pragma unroll
    for (int r = 0; r < 8; ++r) acc[nt].f[r] = 0.0f;

  const int arow = tid >> 1, ac16 = (tid & 1) * 16;  // A: 16 bf16 per thread
  const int brow = tid >> 3, bn8 = (tid & 7) * 8;    // B: 8 bf16 per thread
  const unsigned ldsA = (unsigned)(uintptr_t)&sA[arow][ac16];

  for (int k0 = 0; k0 < K; k0 += 32) {
    __syncthreads();
    // ---- stage A tile 128x32 via async global->LDS (2 x 16B per thread) ----
    {
      const unsigned short* ga = &A[(size_t)(m0 + arow) * K + k0 + ac16];
      async_b128(ldsA, ga);
      async_b128(ldsA + 16u, ga + 8);
    }
    // ---- stage B tile 32x64, transposed into sBt[n][k] ----
    {
      union { uint4 q; unsigned short s[8]; } bv;
      bv.q = *(const uint4*)&B[(size_t)(k0 + brow) * N + n0 + bn8];
      #pragma unroll
      for (int e = 0; e < 8; ++e) sBt[bn8 + e][brow] = bv.s[e];
    }
    if (k0 + 32 < K) {  // prefetch next K tile -> global_prefetch_b8
      __builtin_prefetch(&A[(size_t)(m0 + arow) * K + (k0 + 32) + ac16], 0, 3);
      __builtin_prefetch(&B[(size_t)(k0 + 32 + brow) * N + n0 + bn8], 0, 3);
    }
    wait_async0();
    __syncthreads();

    // ---- fragments in contiguous k order (k = 16*hi + s), same for A and B ----
    FragB a;
    #pragma unroll
    for (int jj = 0; jj < 8; ++jj) {
      const int ka = 16 * hi + 2 * jj;
      a.u[jj] = *(const unsigned int*)&sA[w * 16 + lo][ka];
    }
    static_for<4>([&](auto NT) {
      constexpr int nt = decltype(NT)::value;
      FragB bfr;
      #pragma unroll
      for (int jj = 0; jj < 8; ++jj)
        bfr.u[jj] = *(const unsigned int*)&sBt[nt * 16 + lo][16 * hi + 2 * jj];
      acc[nt].v = __builtin_amdgcn_wmma_f32_16x16x32_bf16(
          false, a.v, false, bfr.v, (short)0, acc[nt].v, nt != 0, false);
    });
  }

  // ---- epilogue: C layout row = r + 8*hi, col = lo ----
  #pragma unroll
  for (int nt = 0; nt < 4; ++nt)
    #pragma unroll
    for (int r = 0; r < 8; ++r) {
      const int m = m0 + w * 16 + r + 8 * hi;
      const int n = n0 + nt * 16 + lo;
      if (Cf) Cf[(size_t)m * N + n] = acc[nt].f[r];
      else    Cb[(size_t)m * N + n] = f32_to_bf16_rn(acc[nt].f[r]);
    }
}

// ---------------------------------------------------------------------------
// Flash-style grouped-query attention (non-causal), bf16 in, bf16 out.
// Q,O: [B, T, 16*64]; K,V: [B, T, 4*64].
// Grid: B * NKV * GRPS * (T/128) blocks; 8 waves, each owns 16 q-rows.
// K chunk staged via async global->LDS; V staged synchronously (transposed).
// ---------------------------------------------------------------------------
__launch_bounds__(256)
__global__ void gqa_flash_bf16(const unsigned short* __restrict__ Q,
                               const unsigned short* __restrict__ Kb,
                               const unsigned short* __restrict__ Vb,
                               unsigned short* __restrict__ O) {
  __shared__ unsigned short sK [64][72];      // [pos][d], 144B rows (16B aligned)
  __shared__ unsigned short sVt[64][72];      // [d][pos]
  __shared__ unsigned short sP [8][16][72];   // per-wave P, [m][pos]

  const int tid = threadIdx.x;
  const int w  = tid >> 5;
  const int l  = tid & 31;
  const int lo = l & 15, hi = l >> 4;

  int bx = blockIdx.x;
  const int qblk = bx & 15; bx >>= 4;
  const int g    = bx & 3;  bx >>= 2;
  const int kv   = bx & 3;  bx >>= 2;
  const int b    = bx;
  const int h    = kv * GRPS + g;
  const int qb0  = qblk * 128 + w * 16;

  // ---- Q A-fragments (16 q-rows x 64 d), kept in registers ----
  FragB aq[2];
  {
    const size_t rowoff = ((size_t)(b * TT + qb0 + lo)) * (NHEADS * DK) + h * DK;
    #pragma unroll
    for (int ks = 0; ks < 2; ++ks)
      #pragma unroll
      for (int jj = 0; jj < 8; ++jj) {
        const int d = 16 * hi + 2 * jj + 32 * ks;
        aq[ks].u[jj] = *(const unsigned int*)&Q[rowoff + d];
      }
  }

  FragC o[4];
  float mrow[8], lrow[8];
  #pragma unroll
  for (int r = 0; r < 8; ++r) { mrow[r] = -3.0e38f; lrow[r] = 0.0f; }
  #pragma unroll
  for (int dt = 0; dt < 4; ++dt)
    #pragma unroll
    for (int r = 0; r < 8; ++r) o[dt].f[r] = 0.0f;

  const int srow = tid >> 2, sc16 = (tid & 3) * 16;  // staging: 16 elems/thread
  const unsigned ldsK = (unsigned)(uintptr_t)&sK[srow][sc16];

  for (int c = 0; c < TT / 64; ++c) {
    __syncthreads();
    // ---- stage K chunk [pos][d] async; V chunk transposed [d][pos] sync ----
    {
      const size_t kvoff =
          ((size_t)(b * TT + c * 64 + srow)) * (NKV * DK) + kv * DK + sc16;
      async_b128(ldsK, &Kb[kvoff]);
      async_b128(ldsK + 16u, &Kb[kvoff + 8]);
      union { uint4 q; unsigned short s[8]; } v0, v1;
      v0.q = *(const uint4*)&Vb[kvoff];
      v1.q = *(const uint4*)&Vb[kvoff + 8];
      #pragma unroll
      for (int e = 0; e < 8; ++e) {
        sVt[sc16 + e][srow]     = v0.s[e];
        sVt[sc16 + 8 + e][srow] = v1.s[e];
      }
    }
    wait_async0();
    __syncthreads();

    // ---- S = (Q * K^T) / sqrt(64): reuse aq[ks] across 4 back-to-back WMMAs
    FragC s4[4];
    #pragma unroll
    for (int nt = 0; nt < 4; ++nt)
      #pragma unroll
      for (int r = 0; r < 8; ++r) s4[nt].f[r] = 0.0f;
    #pragma unroll
    for (int ks = 0; ks < 2; ++ks) {
      static_for<4>([&](auto NT) {
        constexpr int nt = decltype(NT)::value;
        FragB kb;
        #pragma unroll
        for (int jj = 0; jj < 8; ++jj)
          kb.u[jj] = *(const unsigned int*)&sK[nt * 16 + lo][32 * ks + 16 * hi + 2 * jj];
        s4[nt].v = __builtin_amdgcn_wmma_f32_16x16x32_bf16(
            false, aq[ks].v, false, kb.v, (short)0, s4[nt].v, nt != 0, false);
      });
    }
    #pragma unroll
    for (int nt = 0; nt < 4; ++nt)
      #pragma unroll
      for (int r = 0; r < 8; ++r) s4[nt].f[r] *= 0.125f;   // 1/sqrt(DK)

    // ---- online softmax (row reductions stay in 16-lane halves) ----
    float corr[8];
    #pragma unroll
    for (int r = 0; r < 8; ++r) {
      float vmax = fmaxf(fmaxf(s4[0].f[r], s4[1].f[r]),
                         fmaxf(s4[2].f[r], s4[3].f[r]));
      #pragma unroll
      for (int off = 1; off < 16; off <<= 1)
        vmax = fmaxf(vmax, __shfl_xor(vmax, off, 32));
      const float mnew = fmaxf(mrow[r], vmax);
      corr[r] = __expf(mrow[r] - mnew);
      mrow[r] = mnew;
      lrow[r] *= corr[r];
    }
    #pragma unroll
    for (int r = 0; r < 8; ++r) {
      float rs = 0.0f;
      #pragma unroll
      for (int nt = 0; nt < 4; ++nt) {
        const float p = __expf(s4[nt].f[r] - mrow[r]);
        rs += p;
        sP[w][r + 8 * hi][nt * 16 + lo] = f32_to_bf16_rn(p);  // transpose via LDS
      }
      #pragma unroll
      for (int off = 1; off < 16; off <<= 1)
        rs += __shfl_xor(rs, off, 32);
      lrow[r] += rs;
    }
    #pragma unroll
    for (int dt = 0; dt < 4; ++dt)
      #pragma unroll
      for (int r = 0; r < 8; ++r) o[dt].f[r] *= corr[r];
    __syncthreads();

    // ---- O += P * V : one P fragment per ks feeds 4 WMMAs (reuse-A hint) ----
    #pragma unroll
    for (int ks = 0; ks < 2; ++ks) {
      FragB pa;
      #pragma unroll
      for (int jj = 0; jj < 8; ++jj) {
        const int kp = 16 * hi + 2 * jj + 32 * ks;
        pa.u[jj] = *(const unsigned int*)&sP[w][lo][kp];
      }
      static_for<4>([&](auto DT) {
        constexpr int dt = decltype(DT)::value;
        FragB vbf;
        #pragma unroll
        for (int jj = 0; jj < 8; ++jj)
          vbf.u[jj] = *(const unsigned int*)&sVt[dt * 16 + lo][32 * ks + 16 * hi + 2 * jj];
        o[dt].v = __builtin_amdgcn_wmma_f32_16x16x32_bf16(
            false, pa.v, false, vbf.v, (short)0, o[dt].v, dt != 0, false);
      });
    }
  }

  // ---- finalize: O /= l, store bf16 at head h ----
  #pragma unroll
  for (int dt = 0; dt < 4; ++dt)
    #pragma unroll
    for (int r = 0; r < 8; ++r) {
      const float val = o[dt].f[r] / lrow[r];
      const size_t row = (size_t)(b * TT + qb0 + r + 8 * hi);
      O[row * (NHEADS * DK) + h * DK + dt * 16 + lo] = f32_to_bf16_rn(val);
    }
}

// ---------------------------------------------------------------------------
extern "C" void kernel_launch(void* const* d_in, const int* in_sizes, int n_in,
                              void* d_out, int out_size, void* d_ws, size_t ws_size,
                              hipStream_t stream) {
  (void)in_sizes; (void)n_in; (void)out_size; (void)ws_size;
  const float* q  = (const float*)d_in[0];
  const float* k  = (const float*)d_in[1];
  const float* v  = (const float*)d_in[2];
  const float* Wq = (const float*)d_in[3];
  const float* Wk = (const float*)d_in[4];
  const float* Wv = (const float*)d_in[5];
  const float* Wo = (const float*)d_in[6];
  float* out = (float*)d_out;

  char* ws = (char*)d_ws;
  size_t off = 0;
  auto alloc = [&](size_t bytes) -> unsigned short* {
    void* p = ws + off;
    off = (off + bytes + 255) & ~(size_t)255;
    return (unsigned short*)p;
  };
  unsigned short* bq  = alloc((size_t)BTOK * DMODEL * 2);
  unsigned short* bk  = alloc((size_t)BTOK * DMODEL * 2);
  unsigned short* bv  = alloc((size_t)BTOK * DMODEL * 2);
  unsigned short* bWq = alloc((size_t)DMODEL * DMODEL * 2);
  unsigned short* bWk = alloc((size_t)DMODEL * (NKV * DK) * 2);
  unsigned short* bWv = alloc((size_t)DMODEL * (NKV * DK) * 2);
  unsigned short* bWo = alloc((size_t)DMODEL * DMODEL * 2);
  unsigned short* Qp  = alloc((size_t)BTOK * (NHEADS * DK) * 2);
  unsigned short* Kp  = alloc((size_t)BTOK * (NKV * DK) * 2);
  unsigned short* Vp  = alloc((size_t)BTOK * (NKV * DK) * 2);
  unsigned short* AO  = alloc((size_t)BTOK * (NHEADS * DK) * 2);

  auto cvt = [&](const float* src, unsigned short* dst, int n) {
    cvt_f32_to_bf16<<<(n + 255) / 256, 256, 0, stream>>>(src, dst, n);
  };
  cvt(q,  bq,  BTOK * DMODEL);
  cvt(k,  bk,  BTOK * DMODEL);
  cvt(v,  bv,  BTOK * DMODEL);
  cvt(Wq, bWq, DMODEL * DMODEL);
  cvt(Wk, bWk, DMODEL * (NKV * DK));
  cvt(Wv, bWv, DMODEL * (NKV * DK));
  cvt(Wo, bWo, DMODEL * DMODEL);

  dim3 blk(256);
  // Projections (bf16 out)
  gemm_bf16_wmma<<<dim3(DMODEL / 64, BTOK / 128), blk, 0, stream>>>(
      bq, bWq, Qp, nullptr, BTOK, DMODEL, DMODEL);
  gemm_bf16_wmma<<<dim3((NKV * DK) / 64, BTOK / 128), blk, 0, stream>>>(
      bk, bWk, Kp, nullptr, BTOK, NKV * DK, DMODEL);
  gemm_bf16_wmma<<<dim3((NKV * DK) / 64, BTOK / 128), blk, 0, stream>>>(
      bv, bWv, Vp, nullptr, BTOK, NKV * DK, DMODEL);
  // RoPE on Q and K
  rope_bf16<<<(BTOK * NHEADS * 32 + 255) / 256, 256, 0, stream>>>(
      Qp, TT, NHEADS, BTOK * NHEADS * 32);
  rope_bf16<<<(BTOK * NKV * 32 + 255) / 256, 256, 0, stream>>>(
      Kp, TT, NKV, BTOK * NKV * 32);
  // Attention
  gqa_flash_bf16<<<dim3(BB * NKV * GRPS * (TT / 128)), blk, 0, stream>>>(
      Qp, Kp, Vp, AO);
  // Output projection (f32 out -> d_out)
  gemm_bf16_wmma<<<dim3(DMODEL / 64, BTOK / 128), blk, 0, stream>>>(
      AO, bWo, nullptr, out, BTOK, DMODEL, DMODEL);
}
